// NetTCRGlobalMax_18107582120503
// MI455X (gfx1250) — matrix operand order repaired and checked by reading, not compiled, not run on hardware
//
#include <hip/hip_runtime.h>
#include <hip/hip_bf16.h>
#include <math.h>

// ---------------------------------------------------------------------------
// NetTCR on gfx1250: conv-as-GEMM with v_wmma_f32_16x16x32_bf16.
// M = 16-batch tile, N = 16 filters (NF), K = 32 channels (ED).
// A-tiles staged in LDS via async global->LDS copies (ASYNCcnt path).
// ---------------------------------------------------------------------------

typedef __bf16 v16bf __attribute__((ext_vector_type(16)));
typedef __bf16 v8bf  __attribute__((ext_vector_type(8)));
typedef float  v8f   __attribute__((ext_vector_type(8)));
typedef int    v4i   __attribute__((vector_size(16)));
typedef __attribute__((address_space(1))) v4i* gv4i_p;   // global (AS1)
typedef __attribute__((address_space(3))) v4i* lv4i_p;   // LDS (AS3)

#define BSZ   16384
#define ED    32
#define NF    16
#define NIN   7
#define LTOT  85                 // sum of LENS
#define NFRAG 175                // 7 inputs * sum(k) = 7*25 tap fragments
#define LMAX  23
#define WPB   4                  // waves per block (main kernel)

static constexpr int kLens[NIN] = {12, 7, 8, 22, 6, 7, 23};
static constexpr int kPre [NIN] = { 0,12,19, 27,49,55, 62};   // prefix of kLens

// ws layout: [ xT bf16 : BSZ*LTOT*32 ][ wPack bf16 : NFRAG*512 ]
#define XT_ELEMS   ((size_t)BSZ * LTOT * ED)
#define XT_BYTES   (XT_ELEMS * 2)

#if __has_builtin(__builtin_amdgcn_global_load_async_to_lds_b128)
#define HAVE_ASYNC_LDS 1
#else
#define HAVE_ASYNC_LDS 0
#endif

__device__ __forceinline__ void copy16_to_lds(const __bf16* g, __bf16* l) {
#if HAVE_ASYNC_LDS
    __builtin_amdgcn_global_load_async_to_lds_b128((gv4i_p)g, (lv4i_p)l, 0, 0);
#else
    *(v8bf*)l = *(const v8bf*)g;
#endif
}

__device__ __forceinline__ void wait_async_copies() {
#if HAVE_ASYNC_LDS
#if __has_builtin(__builtin_amdgcn_s_wait_asynccnt)
    __builtin_amdgcn_s_wait_asynccnt(0);
#else
    asm volatile("s_wait_asynccnt 0" ::: "memory");
#endif
#endif
}

// ---------------------------------------------------------------------------
// Prep 1: f32 [B, ED, L] -> bf16 [B, L, ED], LDS-staged transpose.
// Each block handles 8 batches (contiguous 8*32*L chunk on both sides).
// ---------------------------------------------------------------------------
__global__ __launch_bounds__(256)
void nettcr_transpose_cvt(const float* __restrict__ src, __bf16* __restrict__ dst, int L) {
    __shared__ __bf16 tile[8 * ED * LMAX];   // max L = 23 -> 11776 B
    const int chunk = ED * L;                // elems per batch
    const int n = 8 * chunk;
    const size_t base = (size_t)blockIdx.x * n;
    for (int idx = threadIdx.x; idx < n; idx += 256) {
        int b = idx / chunk, r = idx - b * chunk;
        int c = r / L, p = r - c * L;
        tile[b * chunk + p * ED + c] = (__bf16)src[base + idx];   // coalesced read
    }
    __syncthreads();
    for (int idx = threadIdx.x; idx < n; idx += 256)
        dst[base + idx] = tile[idx];                              // coalesced write
}

// ---------------------------------------------------------------------------
// Prep 2: pack w{k}[7, NF, ED, k] into per-tap B fragments.
// Fragment (i, kk, d): 32 lanes x 16 bf16.  B layout: lane holds column
// n = lane%16, elements j -> K = 16*(lane/16) + j (channel index).
// KS[kk] = 2kk+1, KOFF[kk] = kk^2.
// ---------------------------------------------------------------------------
__global__ __launch_bounds__(256)
void nettcr_pack_w(const float* __restrict__ w1, const float* __restrict__ w3,
                   const float* __restrict__ w5, const float* __restrict__ w7,
                   const float* __restrict__ w9, __bf16* __restrict__ wP) {
    int idx = blockIdx.x * 256 + threadIdx.x;
    if (idx >= NFRAG * 512) return;
    int fi = idx >> 9, e = idx & 511;
    int lane = e >> 4, j = e & 15;
    int i = fi / 25, r = fi - i * 25;
    int kk = 0;
    #pragma unroll
    for (int t = 1; t < 5; ++t) if (r >= t * t) kk = t;
    int d = r - kk * kk;
    int k = 2 * kk + 1;
    const float* ws = w1;
    if (kk == 1) ws = w3; else if (kk == 2) ws = w5;
    else if (kk == 3) ws = w7; else if (kk == 4) ws = w9;
    int f = lane & 15, half = lane >> 4;
    int c = 16 * half + j;
    wP[idx] = (__bf16)ws[(((i * NF + f) * ED) + c) * k + d];
}

// ---------------------------------------------------------------------------
// Main: one wave per 16-batch tile.  Full net fused.
// Per input i: stage the contiguous 16xLx32 bf16 tile in LDS (async copy),
// then all 5 kernel sizes consume A-fragments from LDS.
// ---------------------------------------------------------------------------
__global__ __launch_bounds__(WPB * 32)
void nettcr_conv_wmma(const __bf16* __restrict__ xT, const __bf16* __restrict__ wP,
                      const float* __restrict__ c1, const float* __restrict__ c3,
                      const float* __restrict__ c5, const float* __restrict__ c7,
                      const float* __restrict__ c9,
                      const float* __restrict__ dw, const float* __restrict__ dbp,
                      float* __restrict__ out) {
    __shared__ __align__(16) __bf16 smem[WPB * 16 * LMAX * ED];   // 94 KB / block

    const int lane = threadIdx.x & 31;
    const int wv   = threadIdx.x >> 5;
    const int tile = blockIdx.x * WPB + wv;
    const int b0   = tile * 16;
    const int lo16 = lane & 15;     // A: batch row within tile; D/B: column n (filter)
    const int half = lane >> 4;

    __bf16* lbase = smem + wv * (16 * LMAX * ED);

    const float* cs[5] = {c1, c3, c5, c7, c9};
    float outacc[8] = {0.f, 0.f, 0.f, 0.f, 0.f, 0.f, 0.f, 0.f};

    #pragma unroll
    for (int i = 0; i < NIN; ++i) {
        const int L = kLens[i];
        // Whole 16-batch tile of input i is contiguous: 16*L*32 bf16.
        const __bf16* gsrc = xT + (size_t)BSZ * ED * kPre[i] + (size_t)b0 * L * ED;
        const int nchunk = 16 * L * ED / 8;        // 16-byte chunks (= 64*L)
        for (int q = lane; q < nchunk; q += 32)    // async global -> LDS, 16B/lane
            copy16_to_lds(gsrc + q * 8, lbase + q * 8);
        wait_async_copies();

        #pragma unroll
        for (int kk = 0; kk < 5; ++kk) {
            const int k = 2 * kk + 1, pad = kk, koff = kk * kk;
            v16bf bfr[9];
            #pragma unroll
            for (int d = 0; d < k; ++d)
                bfr[d] = *(const v16bf*)(wP + ((size_t)((i * 25 + koff + d) * 32 + lane)) * 16);
            const float bias = cs[kk][i * NF + lo16];
            const float wd   = dw[(i * 5 + kk) * NF + lo16];
            v8f maxv = {0.f, 0.f, 0.f, 0.f, 0.f, 0.f, 0.f, 0.f};  // relu >= 0
            for (int t = 0; t < L; ++t) {
                v8f acc = {0.f, 0.f, 0.f, 0.f, 0.f, 0.f, 0.f, 0.f};
                #pragma unroll
                for (int d = 0; d < k; ++d) {
                    const int p = t + d - pad;        // wave-uniform: scalar branch
                    if (p >= 0 && p < L) {
                        // A layout: lane holds M=lane%16, K=[16h..16h+15]
                        const __bf16* ap = lbase + (lo16 * L + p) * ED + 16 * half;
                        v8bf alo = *(const v8bf*)ap;            // ds_load_b128
                        v8bf ahi = *(const v8bf*)(ap + 8);      // ds_load_b128
                        v16bf a = __builtin_shufflevector(alo, ahi,
                                   0,1,2,3,4,5,6,7,8,9,10,11,12,13,14,15);
                        acc = __builtin_amdgcn_wmma_f32_16x16x32_bf16(
                                  false, a, false, bfr[d], (short)0, acc, false, false);
                    }
                }
                #pragma unroll
                for (int rr = 0; rr < 8; ++rr) {
                    float v = acc[rr] + bias;
                    v = v > 0.f ? v : 0.f;
                    maxv[rr] = v > maxv[rr] ? v : maxv[rr];
                }
            }
            #pragma unroll
            for (int rr = 0; rr < 8; ++rr) outacc[rr] += maxv[rr] * wd;
        }
    }

    // Reduce over the 16 filter columns within each half-group of lanes.
    #pragma unroll
    for (int rr = 0; rr < 8; ++rr) {
        float v = outacc[rr];
        v += __shfl_xor(v, 1, 32);
        v += __shfl_xor(v, 2, 32);
        v += __shfl_xor(v, 4, 32);
        v += __shfl_xor(v, 8, 32);
        outacc[rr] = v;
    }
    if (lo16 == 0) {                       // lanes 0 (rows 0-7) and 16 (rows 8-15)
        const float bb = dbp[0];
        #pragma unroll
        for (int rr = 0; rr < 8; ++rr) {
            float x = outacc[rr] + bb;
            out[b0 + half * 8 + rr] = 1.0f / (1.0f + __expf(-x));
        }
    }
}

// ---------------------------------------------------------------------------
extern "C" void kernel_launch(void* const* d_in, const int* in_sizes, int n_in,
                              void* d_out, int out_size, void* d_ws, size_t ws_size,
                              hipStream_t stream) {
    (void)in_sizes; (void)n_in; (void)out_size; (void)ws_size;
    // dict order: pep,a1,a2,a3,b1,b2,b3, w1,c1,w3,c3,w5,c5,w7,c7,w9,c9, dw,db
    const float* x[NIN];
    for (int i = 0; i < NIN; ++i) x[i] = (const float*)d_in[i];
    const float* w1 = (const float*)d_in[7];
    const float* c1 = (const float*)d_in[8];
    const float* w3 = (const float*)d_in[9];
    const float* c3 = (const float*)d_in[10];
    const float* w5 = (const float*)d_in[11];
    const float* c5 = (const float*)d_in[12];
    const float* w7 = (const float*)d_in[13];
    const float* c7 = (const float*)d_in[14];
    const float* w9 = (const float*)d_in[15];
    const float* c9 = (const float*)d_in[16];
    const float* dw = (const float*)d_in[17];
    const float* db = (const float*)d_in[18];
    float* out = (float*)d_out;

    __bf16* xT = (__bf16*)d_ws;
    __bf16* wP = (__bf16*)((char*)d_ws + XT_BYTES);

    for (int i = 0; i < NIN; ++i)
        nettcr_transpose_cvt<<<BSZ / 8, 256, 0, stream>>>(
            x[i], xT + (size_t)BSZ * ED * kPre[i], kLens[i]);

    nettcr_pack_w<<<(NFRAG * 512 + 255) / 256, 256, 0, stream>>>(w1, w3, w5, w7, w9, wP);

    nettcr_conv_wmma<<<(BSZ / 16) / WPB, WPB * 32, 0, stream>>>(
        xT, wP, c1, c3, c5, c7, c9, dw, db, out);
}
